// AttnBlock_21698174779586
// MI455X (gfx1250) — matrix operand order repaired and checked by reading, not compiled, not run on hardware
//
#include <hip/hip_runtime.h>

// ---------------------------------------------------------------------------
// AttnBlock for MI455X (gfx1250): GroupNorm + QKV proj + softmax-attention +
// output proj + residual.  All GEMMs use v_wmma_f32_16x16x32_bf16 (wave32).
// ---------------------------------------------------------------------------

typedef __bf16 bf16;
typedef __bf16 bf16x8  __attribute__((ext_vector_type(8)));
typedef __bf16 bf16x16 __attribute__((ext_vector_type(16)));
typedef float  v8f     __attribute__((ext_vector_type(8)));

#define C_DIM 512
#define N_DIM 1024
#define B_DIM 16
#define INV_SQRT_C 0.04419417382415922f   // 1/sqrt(512)

// ---- WMMA helpers ---------------------------------------------------------

__device__ __forceinline__ v8f wmma_bf16(bf16x16 a, bf16x16 b, v8f c) {
  // D = A(16x32) * B(32x16) + C, fp32 accumulate
  return __builtin_amdgcn_wmma_f32_16x16x32_bf16(
      /*neg_a=*/false, a, /*neg_b=*/false, b,
      /*c_mod=*/(short)0, c, /*reuse_a=*/false, /*reuse_b=*/false);
}

// A operand: 16x32 (MxK) tile, element (m,k) at p0[m*ld + k] (K-contiguous rows).
// Per-lane: row = lane&15; K chunks {koff..koff+7} and {16+koff..23+koff}, koff = lane<16?0:8.
__device__ __forceinline__ bf16x16 load_a_tile(const bf16* p0, int ld) {
  const int lane = threadIdx.x & 31;
  const int mrow = lane & 15;
  const int koff = (lane < 16) ? 0 : 8;
  const bf16* p = p0 + mrow * ld + koff;
  bf16x8 lo = *(const bf16x8*)(p);
  bf16x8 hi = *(const bf16x8*)(p + 16);
  return __builtin_shufflevector(lo, hi, 0,1,2,3,4,5,6,7,8,9,10,11,12,13,14,15);
}

// B operand: 32x16 (KxN) tile, element (k,n) at p0[n*ld + k] (K-contiguous per column,
// i.e. source stored N-major / "B-transposed").  Per-lane: col = lane&15;
// K chunks {kb..kb+7} and {16+kb..23+kb}, kb = lane<16 ? 0 : 8.
__device__ __forceinline__ bf16x16 load_bt_tile(const bf16* p0, int ld) {
  const int lane = threadIdx.x & 31;
  const int ncol = lane & 15;
  const int kb   = (lane < 16) ? 0 : 8;
  const bf16* p = p0 + ncol * ld + kb;
  bf16x8 lo = *(const bf16x8*)(p);
  bf16x8 hi = *(const bf16x8*)(p + 16);
  return __builtin_shufflevector(lo, hi, 0,1,2,3,4,5,6,7,8,9,10,11,12,13,14,15);
}

// ---- Kernel 1: fp32 -> bf16 weight conversion -----------------------------

__global__ __launch_bounds__(256) void cvt_bf16_kernel(const float* __restrict__ src,
                                                       bf16* __restrict__ dst) {
  int i = blockIdx.x * 256 + threadIdx.x;
  dst[i] = (bf16)src[i];
}

// ---- Kernel 2: GroupNorm, writes hT [b][n][c] (c contiguous) --------------

__global__ __launch_bounds__(256) void gn_kernel(const float* __restrict__ x,
                                                 const float* __restrict__ gscale,
                                                 const float* __restrict__ gbias,
                                                 bf16* __restrict__ hT) {
  const int b = blockIdx.x >> 5;
  const int g = blockIdx.x & 31;
  const int tid = threadIdx.x;
  const float* xp = x + ((b * C_DIM + g * 16) << 10);   // 16 channels x 1024

  float s = 0.f, ss = 0.f;
  for (int i = tid; i < 16 * N_DIM; i += 256) {
    float v = xp[i];
    s += v; ss += v * v;
  }
  __shared__ float rs[256], rss[256];
  rs[tid] = s; rss[tid] = ss;
  __syncthreads();
  for (int off = 128; off > 0; off >>= 1) {
    if (tid < off) { rs[tid] += rs[tid + off]; rss[tid] += rss[tid + off]; }
    __syncthreads();
  }
  const float inv_n = 1.0f / (16.0f * N_DIM);
  float mu   = rs[0] * inv_n;
  float var  = rss[0] * inv_n - mu * mu;
  float rstd = rsqrtf(var + 1e-5f);

  for (int i = tid; i < 16 * N_DIM; i += 256) {
    int cg = i >> 10;
    int n  = i & (N_DIM - 1);
    int c  = g * 16 + cg;
    float v = (xp[i] - mu) * rstd * gscale[c] + gbias[c];
    hT[((b << 10) + n) * C_DIM + c] = (bf16)v;
  }
}

// ---- Kernel 3: QKV projections --------------------------------------------
// grid (16, 4, B*3); 8 waves; wave w owns o-tile (by*8+w), 4 n-tiles each.
// q,k written transposed [b][n][c] (single 16B store/lane); v natural [b][c][n].
// q pre-scaled by 1/sqrt(C).

__global__ __launch_bounds__(256) void qkv_kernel(const bf16* __restrict__ hT,
                                                  const bf16* __restrict__ wqb,
                                                  const bf16* __restrict__ wkb,
                                                  const bf16* __restrict__ wvb,
                                                  const float* __restrict__ bq,
                                                  const float* __restrict__ bk,
                                                  const float* __restrict__ bv,
                                                  bf16* __restrict__ qT,
                                                  bf16* __restrict__ kT,
                                                  bf16* __restrict__ vO) {
  const int b = blockIdx.z / 3;
  const int z = blockIdx.z % 3;
  const int wave = threadIdx.x >> 5;
  const int lane = threadIdx.x & 31;
  const int o0 = (blockIdx.y * 8 + wave) * 16;

  const bf16*  W    = (z == 0) ? wqb : (z == 1) ? wkb : wvb;
  const float* bias = (z == 0) ? bq  : (z == 1) ? bk  : bv;
  const bf16*  Bsrc = hT + (b << 10) * C_DIM;    // [n][c], ld = 512
  const float  scl  = (z == 0) ? INV_SQRT_C : 1.0f;

  v8f acc[4] = {};
  for (int c0 = 0; c0 < C_DIM; c0 += 32) {
    bf16x16 a = load_a_tile(W + o0 * C_DIM + c0, C_DIM);
    for (int j = 0; j < 4; ++j) {
      int n0 = (blockIdx.x * 4 + j) * 16;
      bf16x16 bb = load_bt_tile(Bsrc + n0 * C_DIM + c0, C_DIM);
      acc[j] = wmma_bf16(a, bb, acc[j]);
    }
  }

  const int ncol = lane & 15;
  const int roff = (lane < 16) ? 0 : 8;
  if (z < 2) {
    bf16* dst = (z == 0) ? qT : kT;
    for (int j = 0; j < 4; ++j) {
      int n0 = (blockIdx.x * 4 + j) * 16;
      bf16x8 st;
      for (int r = 0; r < 8; ++r)
        st[r] = (bf16)((acc[j][r] + bias[o0 + roff + r]) * scl);
      *(bf16x8*)(dst + ((b << 10) + n0 + ncol) * C_DIM + o0 + roff) = st;
    }
  } else {
    for (int j = 0; j < 4; ++j) {
      int n0 = (blockIdx.x * 4 + j) * 16;
      for (int r = 0; r < 8; ++r) {
        int o = o0 + roff + r;
        vO[((b * C_DIM + o) << 10) + n0 + ncol] = (bf16)(acc[j][r] + bias[o]);
      }
    }
  }
}

// ---- Kernel 4: attention (S = q^T k, softmax, HV = V P^T) -----------------
// grid (64, B); one workgroup = one (batch, 16-row block).  S fp32 + P bf16 in LDS.

#define S_STRIDE 1028   // 16*1028*4  = 65792 B
#define P_STRIDE 1032   // 16*1032*2  = 33024 B

__global__ __launch_bounds__(256) void attn_kernel(const bf16* __restrict__ qT,
                                                   const bf16* __restrict__ kT,
                                                   const bf16* __restrict__ vO,
                                                   bf16* __restrict__ hvT) {
  __shared__ __align__(16) float Ssh[16 * S_STRIDE];
  __shared__ __align__(16) bf16  Psh[16 * P_STRIDE];
  __shared__ float red[16][17];
  __shared__ float rowmax[16];
  __shared__ float rowrsum[16];

  const int b    = blockIdx.y;
  const int n0   = blockIdx.x * 16;
  const int tid  = threadIdx.x;
  const int wave = tid >> 5;
  const int lane = tid & 31;

  const bf16* Qb = qT + ((b << 10) + n0) * C_DIM;  // [16][512], already * 1/sqrt(C)
  const bf16* Kb = kT + (b << 10) * C_DIM;         // [1024][512]

  // ---- Phase 1: S[16][1024] = q^T k  (each wave: 8 m-tiles) ----
  {
    v8f acc[8] = {};
    for (int c0 = 0; c0 < C_DIM; c0 += 32) {
      bf16x16 a = load_a_tile(Qb + c0, C_DIM);            // shared across 8 tiles
      for (int j = 0; j < 8; ++j) {
        int m0 = (wave + (j << 3)) * 16;
        bf16x16 bb = load_bt_tile(Kb + m0 * C_DIM + c0, C_DIM);
        acc[j] = wmma_bf16(a, bb, acc[j]);
      }
    }
    const int ncol = lane & 15;
    const int roff = (lane < 16) ? 0 : 8;
    for (int j = 0; j < 8; ++j) {
      int m0 = (wave + (j << 3)) * 16;
      for (int r = 0; r < 8; ++r)
        Ssh[(roff + r) * S_STRIDE + m0 + ncol] = acc[j][r];
    }
  }
  __syncthreads();

  // ---- Phase 2: row softmax (16 threads per row, 64 strided elems each) ----
  {
    const int r   = tid & 15;
    const int seg = tid >> 4;
    float mx = -1e30f;
    for (int j = 0; j < 64; ++j)
      mx = fmaxf(mx, Ssh[r * S_STRIDE + seg + (j << 4)]);
    red[r][seg] = mx;
    __syncthreads();
    if (seg == 0) {
      float m2 = red[r][0];
      for (int t = 1; t < 16; ++t) m2 = fmaxf(m2, red[r][t]);
      rowmax[r] = m2;
    }
    __syncthreads();
    const float rm = rowmax[r];
    float sum = 0.f;
    for (int j = 0; j < 64; ++j) {
      int m = seg + (j << 4);
      float e = __expf(Ssh[r * S_STRIDE + m] - rm);
      Psh[r * P_STRIDE + m] = (bf16)e;
      sum += e;
    }
    red[r][seg] = sum;
    __syncthreads();
    if (seg == 0) {
      float s2 = 0.f;
      for (int t = 0; t < 16; ++t) s2 += red[r][t];
      rowrsum[r] = 1.0f / s2;
    }
    __syncthreads();
    const float rinv = rowrsum[r];
    for (int j = 0; j < 64; ++j) {
      int m = seg + (j << 4);
      Psh[r * P_STRIDE + m] = (bf16)((float)Psh[r * P_STRIDE + m] * rinv);
    }
  }
  __syncthreads();

  // ---- Phase 3: HV = V(512x1024) * P^T(1024x16); each wave: 4 c-tiles ----
  {
    const bf16* Vbase = vO + b * C_DIM * N_DIM;   // [512][1024]
    v8f acc[4] = {};
    for (int m0 = 0; m0 < N_DIM; m0 += 32) {
      bf16x16 bb = load_bt_tile(Psh + m0, P_STRIDE);     // ds loads from LDS
      for (int j = 0; j < 4; ++j) {
        int c0 = (wave + (j << 3)) * 16;
        bf16x16 a = load_a_tile(Vbase + c0 * N_DIM + m0, N_DIM);
        acc[j] = wmma_bf16(a, bb, acc[j]);
      }
    }
    const int ncol = lane & 15;
    const int roff = (lane < 16) ? 0 : 8;
    for (int j = 0; j < 4; ++j) {
      int c0 = (wave + (j << 3)) * 16;
      bf16x8 st;
      for (int r = 0; r < 8; ++r) st[r] = (bf16)acc[j][r];
      *(bf16x8*)(hvT + ((b << 10) + n0 + ncol) * C_DIM + c0 + roff) = st;
    }
  }
}

// ---- Kernel 5: output projection + bias + residual, fp32 out --------------
// grid (16, 4, B)

__global__ __launch_bounds__(256) void proj_kernel(const bf16* __restrict__ hvT,
                                                   const bf16* __restrict__ wob,
                                                   const float* __restrict__ bo,
                                                   const float* __restrict__ x,
                                                   float* __restrict__ out) {
  const int b = blockIdx.z;
  const int wave = threadIdx.x >> 5;
  const int lane = threadIdx.x & 31;
  const int o0 = (blockIdx.y * 8 + wave) * 16;
  const bf16* Bsrc = hvT + (b << 10) * C_DIM;   // [n][c]

  v8f acc[4] = {};
  for (int c0 = 0; c0 < C_DIM; c0 += 32) {
    bf16x16 a = load_a_tile(wob + o0 * C_DIM + c0, C_DIM);
    for (int j = 0; j < 4; ++j) {
      int n0 = (blockIdx.x * 4 + j) * 16;
      bf16x16 bb = load_bt_tile(Bsrc + n0 * C_DIM + c0, C_DIM);
      acc[j] = wmma_bf16(a, bb, acc[j]);
    }
  }
  const int ncol = lane & 15;
  const int roff = (lane < 16) ? 0 : 8;
  for (int j = 0; j < 4; ++j) {
    int n0 = (blockIdx.x * 4 + j) * 16;
    for (int r = 0; r < 8; ++r) {
      int o = o0 + roff + r;
      int idx = ((b * C_DIM + o) << 10) + n0 + ncol;
      out[idx] = x[idx] + bo[o] + acc[j][r];
    }
  }
}

// ---------------------------------------------------------------------------

extern "C" void kernel_launch(void* const* d_in, const int* in_sizes, int n_in,
                              void* d_out, int out_size, void* d_ws, size_t ws_size,
                              hipStream_t stream) {
  (void)in_sizes; (void)n_in; (void)out_size; (void)ws_size;

  const float* x    = (const float*)d_in[0];
  const float* gsc  = (const float*)d_in[1];
  const float* gbi  = (const float*)d_in[2];
  const float* wq   = (const float*)d_in[3];
  const float* bq   = (const float*)d_in[4];
  const float* wk   = (const float*)d_in[5];
  const float* bk   = (const float*)d_in[6];
  const float* wv   = (const float*)d_in[7];
  const float* bv   = (const float*)d_in[8];
  const float* wo   = (const float*)d_in[9];
  const float* bo   = (const float*)d_in[10];
  float* out = (float*)d_out;

  const size_t TEN = (size_t)B_DIM * N_DIM * C_DIM * sizeof(bf16);  // 16 MiB
  char* ws = (char*)d_ws;
  bf16* hT  = (bf16*)(ws);             // aliased: reused as hvT after QKV
  bf16* qT  = (bf16*)(ws + TEN);
  bf16* kT  = (bf16*)(ws + 2 * TEN);
  bf16* vO  = (bf16*)(ws + 3 * TEN);
  bf16* wqb = (bf16*)(ws + 4 * TEN);
  bf16* wkb = wqb + C_DIM * C_DIM;
  bf16* wvb = wkb + C_DIM * C_DIM;
  bf16* wob = wvb + C_DIM * C_DIM;
  bf16* hvT = hT;

  dim3 blk(256);
  // weights -> bf16
  cvt_bf16_kernel<<<dim3(C_DIM * C_DIM / 256), blk, 0, stream>>>(wq, wqb);
  cvt_bf16_kernel<<<dim3(C_DIM * C_DIM / 256), blk, 0, stream>>>(wk, wkb);
  cvt_bf16_kernel<<<dim3(C_DIM * C_DIM / 256), blk, 0, stream>>>(wv, wvb);
  cvt_bf16_kernel<<<dim3(C_DIM * C_DIM / 256), blk, 0, stream>>>(wo, wob);
  // group norm -> hT
  gn_kernel<<<dim3(B_DIM * 32), blk, 0, stream>>>(x, gsc, gbi, hT);
  // qkv projections
  qkv_kernel<<<dim3(16, 4, B_DIM * 3), blk, 0, stream>>>(hT, wqb, wkb, wvb,
                                                         bq, bk, bv, qT, kT, vO);
  // attention
  attn_kernel<<<dim3(64, B_DIM), blk, 0, stream>>>(qT, kT, vO, hvT);
  // output projection + residual
  proj_kernel<<<dim3(16, 4, B_DIM), blk, 0, stream>>>(hvT, wob, bo, x, out);
}
